// DeformableWindowAttention_53738630807644
// MI455X (gfx1250) — matrix operand order repaired and checked by reading, not compiled, not run on hardware
//
#include <hip/hip_runtime.h>
#include <hip/hip_bf16.h>

// ---------------------------------------------------------------------------
// DeformableWindowAttention for MI455X (gfx1250), wave32 + WMMA f16.
// B=8, DIM=64, HEADS=4, HD=16, H=W=256, WS=AWS=8  -> no padding needed.
// All heavy matmuls go through v_wmma_f32_16x16x32_f16.
// ---------------------------------------------------------------------------

typedef __attribute__((ext_vector_type(16))) _Float16 v16h;
typedef __attribute__((ext_vector_type(8)))  float    v8f;

#define NPIX   65536        // 256*256 pixels per (b, channel) plane
#define NB     8
#define DIM    64
#define HEADS  4
#define HD     16
#define WN     32           // windows per side
#define SCALE  0.25f        // HD^-0.5

__device__ __forceinline__ v8f wmma_f16(v16h a, v16h b, v8f c) {
    return __builtin_amdgcn_wmma_f32_16x16x32_f16(
        /*neg_a=*/false, a, /*neg_b=*/false, b,
        /*c_mod=*/(short)0, c, /*reuse_a=*/false, /*reuse_b=*/false);
}

// A-matrix 16x32 f16, row-major source A[m][k] = base[m*ldm + k].
// Layout: lane<16 holds row m=lane, K=0..15;
// lane>=16 holds row m=lane-16, K=16..31.
__device__ __forceinline__ v16h load_A(const _Float16* base, int ldm, int lane, int kmax) {
    int m = lane & 15, half = lane >> 4;
    v16h a;
#pragma unroll
    for (int v = 0; v < 8; ++v) {
        int k = 2 * v + (half ? 16 : 0);
        a[2 * v]     = (k     < kmax) ? base[m * ldm + k]     : (_Float16)0.0f;
        a[2 * v + 1] = (k + 1 < kmax) ? base[m * ldm + k + 1] : (_Float16)0.0f;
    }
    return a;
}

// B-matrix 32x16 f16, row-major source B[k][n] = base[k*ldk + n].
// lane<16: n=lane, K pairs {2v,2v+1}; lane>=16: n=lane-16, K pairs {16+2v,...}.
__device__ __forceinline__ v16h load_B(const _Float16* base, int ldk, int lane) {
    int n = lane & 15, half = lane >> 4;
    v16h b;
#pragma unroll
    for (int v = 0; v < 8; ++v) {
        int k = 2 * v + (half ? 16 : 0);
        b[2 * v]     = base[k * ldk + n];
        b[2 * v + 1] = base[(k + 1) * ldk + n];
    }
    return b;
}

// B-matrix from a transposed source: B[k][n] = base[n*ldn + k], zero pad k>=kmax.
__device__ __forceinline__ v16h load_BT(const _Float16* base, int ldn, int lane, int kmax) {
    int n = lane & 15, half = lane >> 4;
    v16h b;
#pragma unroll
    for (int v = 0; v < 8; ++v) {
        int k = 2 * v + (half ? 16 : 0);
        b[2 * v]     = (k     < kmax) ? base[n * ldn + k]     : (_Float16)0.0f;
        b[2 * v + 1] = (k + 1 < kmax) ? base[n * ldn + k + 1] : (_Float16)0.0f;
    }
    return b;
}

// ---------------------------------------------------------------------------
// Kernel 1: 8x8 avg-pool -> leaky_relu -> off/sc heads (64->8 each).
// One block per (b, wh, ww); 64 threads (one per channel).
// off layout: offbuf[((b*4+head)*2 + ch)*1024 + wh*32 + ww], ch0 = x.
// ---------------------------------------------------------------------------
__global__ void __launch_bounds__(64)
pool_off_kernel(const float* __restrict__ x,
                const float* __restrict__ off_w, const float* __restrict__ off_b,
                const float* __restrict__ sc_w,  const float* __restrict__ sc_b,
                float* __restrict__ offbuf, float* __restrict__ scbuf) {
    __shared__ float act[64];
    int bid = blockIdx.x;
    int b  = bid >> 10;
    int wh = (bid >> 5) & 31;
    int ww = bid & 31;
    int c  = threadIdx.x;

    const float* plane = x + ((size_t)(b * DIM + c)) * NPIX + (size_t)(wh * 8) * 256 + ww * 8;
    float s = 0.0f;
#pragma unroll
    for (int i = 0; i < 8; ++i)
#pragma unroll
        for (int j = 0; j < 8; ++j)
            s += plane[i * 256 + j];
    s *= (1.0f / 64.0f);
    act[c] = (s > 0.0f) ? s : 0.01f * s;    // leaky_relu
    __syncthreads();

    if (c < 16) {
        bool isOff = (c < 8);
        int  o     = isOff ? c : (c - 8);
        const float* wr = (isOff ? off_w : sc_w) + o * DIM;
        float d = isOff ? off_b[o] : sc_b[o];
#pragma unroll
        for (int k = 0; k < DIM; ++k) d += wr[k] * act[k];
        int head = o >> 1, ch = o & 1;
        int idx  = ((b * HEADS + head) * 2 + ch) * 1024 + wh * 32 + ww;
        if (isOff) offbuf[idx] = d * (1.0f / 32.0f);   // * 1/(H/WS)
        else       scbuf[idx]  = d;
    }
}

// ---------------------------------------------------------------------------
// Kernel 2: QKV projection GEMM. (524288 pix x 64) @ (64 x 192) -> f16 planes.
// Block = 256 threads = 8 waves; wave w computes a 16(pix) x 16(outch) tile.
// grid = (4096, 12).  Output layout: qkv[(b*192 + o)*NPIX + pix].
// ---------------------------------------------------------------------------
__global__ void __launch_bounds__(256)
qkv_gemm_kernel(const float* __restrict__ x,
                const float* __restrict__ w,
                const float* __restrict__ bias,
                _Float16* __restrict__ out) {
    __shared__ _Float16 Bsh[64][16];        // B[k][n]
    __shared__ _Float16 Ash[8][16][64];     // per wave: A[m][k]
    int lane = threadIdx.x & 31;
    int wave = threadIdx.x >> 5;
    int o0   = blockIdx.y * 16;

    // stage B: w[(o0+n)*64 + k] (whole block)
    for (int idx = threadIdx.x; idx < 1024; idx += 256) {
        int k = idx >> 4, n = idx & 15;
        Bsh[k][n] = (_Float16)w[(o0 + n) * DIM + k];
    }

    // stage A (per wave): 16 consecutive pixels x 64 channels
    long pix0 = (long)blockIdx.x * 128 + wave * 16;
    int  b    = (int)(pix0 >> 16);
    int  p0   = (int)(pix0 & (NPIX - 1));
    const float* xb = x + (size_t)b * DIM * NPIX;
    __builtin_prefetch(xb + p0 + 16, 0, 0);   // global_prefetch_b8 of next tile
    for (int idx = lane; idx < 1024; idx += 32) {
        int c = idx >> 4, m = idx & 15;
        Ash[wave][m][c] = (_Float16)xb[(size_t)c * NPIX + p0 + m];
    }
    __syncthreads();

    v8f acc = {};
    acc = wmma_f16(load_A(&Ash[wave][0][0],  64, lane, 32),
                   load_B(&Bsh[0][0],  16, lane), acc);
    acc = wmma_f16(load_A(&Ash[wave][0][32], 64, lane, 32),
                   load_B(&Bsh[32][0], 16, lane), acc);

    int   n    = lane & 15, half = lane >> 4;
    float bn   = bias[o0 + n];
    _Float16* outp = out + ((size_t)b * 192 + o0 + n) * NPIX;
#pragma unroll
    for (int j = 0; j < 8; ++j) {
        int m = j + (half ? 8 : 0);
        outp[p0 + m] = (_Float16)(acc[j] + bn);
    }
}

// ---------------------------------------------------------------------------
// Kernel 3: per (window, head): gather k/v by bilinear grid-sample (zeros pad),
// dots = q k^T * SCALE + rel_bias, softmax, out = attn v.  One wave / block.
// blockIdx.x = ((b*4+head)<<10) + wh*32 + ww ; 32768 blocks of 32 threads.
// ---------------------------------------------------------------------------
__global__ void __launch_bounds__(32)
attn_kernel(const _Float16* __restrict__ qkv,
            const float* __restrict__ offbuf, const float* __restrict__ scbuf,
            const float* __restrict__ rel_bias,
            _Float16* __restrict__ outbuf) {
    __shared__ _Float16 qLds[64][16];
    __shared__ _Float16 kLds[64][16];
    __shared__ _Float16 vLds[64][16];
    __shared__ float    dotsLds[64][64];
    __shared__ _Float16 attnLds[64][64];

    int lane = threadIdx.x;
    int win  = blockIdx.x;
    int bh   = win >> 10;           // b*4 + head
    int wh   = (win >> 5) & 31;
    int ww   = win & 31;
    int b    = bh >> 2;
    int head = bh & 3;
    int y0   = wh * 8, x0 = ww * 8;
    int widx = wh * 32 + ww;

    float offx = offbuf[(bh * 2 + 0) * 1024 + widx];
    float offy = offbuf[(bh * 2 + 1) * 1024 + widx];
    float scx  = scbuf [(bh * 2 + 0) * 1024 + widx];
    float scy  = scbuf [(bh * 2 + 1) * 1024 + widx];

    const _Float16* qpl = qkv + ((size_t)b * 192 +   0 + head * HD) * NPIX;
    const _Float16* kpl = qkv + ((size_t)b * 192 +  64 + head * HD) * NPIX;
    const _Float16* vpl = qkv + ((size_t)b * 192 + 128 + head * HD) * NPIX;

    // ---- load q tile (64 positions x 16 dims) ----
    for (int idx = lane; idx < 1024; idx += 32) {
        int m = idx >> 4, d = idx & 15;
        qLds[m][d] = qpl[(size_t)d * NPIX + (size_t)(y0 + (m >> 3)) * 256 + x0 + (m & 7)];
    }

    // ---- bilinear gather of k/v (grid_sample, zeros padding) ----
    for (int p = lane; p < 64; p += 32) {
        int i = p >> 3, j = p & 7;
        float gxn = (-1.0f + (2.0f / 255.0f) * (float)(x0 + j))
                  + ((2.0f / 255.0f) * (float)j - (7.0f / 255.0f)) * scx + offx;
        float gyn = (-1.0f + (2.0f / 255.0f) * (float)(y0 + i))
                  + ((2.0f / 255.0f) * (float)i - (7.0f / 255.0f)) * scy + offy;
        float gx = (gxn + 1.0f) * 0.5f * 255.0f;
        float gy = (gyn + 1.0f) * 0.5f * 255.0f;
        int ix0 = (int)floorf(gx), iy0 = (int)floorf(gy);
        int ix1 = ix0 + 1,          iy1 = iy0 + 1;
        float fx = gx - (float)ix0, fy = gy - (float)iy0;
        float w00 = (1.0f - fy) * (1.0f - fx), w01 = (1.0f - fy) * fx;
        float w10 = fy * (1.0f - fx),          w11 = fy * fx;
        bool vx0 = (ix0 >= 0) & (ix0 < 256), vx1 = (ix1 >= 0) & (ix1 < 256);
        bool vy0 = (iy0 >= 0) & (iy0 < 256), vy1 = (iy1 >= 0) & (iy1 < 256);
        float m00 = (vy0 && vx0) ? w00 : 0.0f, m01 = (vy0 && vx1) ? w01 : 0.0f;
        float m10 = (vy1 && vx0) ? w10 : 0.0f, m11 = (vy1 && vx1) ? w11 : 0.0f;
        int cx0 = min(max(ix0, 0), 255), cx1 = min(max(ix1, 0), 255);
        int cy0 = min(max(iy0, 0), 255), cy1 = min(max(iy1, 0), 255);
        int a00 = cy0 * 256 + cx0, a01 = cy0 * 256 + cx1;
        int a10 = cy1 * 256 + cx0, a11 = cy1 * 256 + cx1;
#pragma unroll
        for (int d = 0; d < HD; ++d) {
            const _Float16* kp = kpl + (size_t)d * NPIX;
            const _Float16* vp = vpl + (size_t)d * NPIX;
            float kv = m00 * (float)kp[a00] + m01 * (float)kp[a01]
                     + m10 * (float)kp[a10] + m11 * (float)kp[a11];
            float vv = m00 * (float)vp[a00] + m01 * (float)vp[a01]
                     + m10 * (float)vp[a10] + m11 * (float)vp[a11];
            kLds[p][d] = (_Float16)kv;
            vLds[p][d] = (_Float16)vv;
        }
    }
    __syncthreads();

    // ---- dots = q k^T * SCALE + rel_bias  (16 WMMAs, K=16 zero-padded) ----
    int n = lane & 15, half = lane >> 4;
#pragma unroll
    for (int qt = 0; qt < 4; ++qt) {
#pragma unroll
        for (int kt = 0; kt < 4; ++kt) {
            v8f c = {};
            c = wmma_f16(load_A (&qLds[qt * 16][0], 16, lane, 16),
                         load_BT(&kLds[kt * 16][0], 16, lane, 16), c);
            int kk = kt * 16 + n;
            int ki = kk >> 3, kj = kk & 7;
#pragma unroll
            for (int j0 = 0; j0 < 8; ++j0) {
                int m  = j0 + (half ? 8 : 0);
                int qq = qt * 16 + m;
                int qi = qq >> 3, qj = qq & 7;
                int ridx = (qi - ki + 7) * 15 + (qj - kj + 7);
                dotsLds[qq][kk] = c[j0] * SCALE + rel_bias[ridx * HEADS + head];
            }
        }
    }
    __syncthreads();

    // ---- softmax over keys (2 rows per lane) ----
    for (int r = lane; r < 64; r += 32) {
        float mx = -1e30f;
        for (int kc = 0; kc < 64; ++kc) mx = fmaxf(mx, dotsLds[r][kc]);
        float sum = 0.0f;
        for (int kc = 0; kc < 64; ++kc) {
            float e = __expf(dotsLds[r][kc] - mx);
            dotsLds[r][kc] = e;
            sum += e;
        }
        float inv = 1.0f / sum;
        for (int kc = 0; kc < 64; ++kc)
            attnLds[r][kc] = (_Float16)(dotsLds[r][kc] * inv);
    }
    __syncthreads();

    // ---- out = attn @ v  (4 M-tiles x 2 K-chunks) ----
#pragma unroll
    for (int mt = 0; mt < 4; ++mt) {
        v8f c = {};
#pragma unroll
        for (int kc = 0; kc < 2; ++kc) {
            c = wmma_f16(load_A(&attnLds[mt * 16][kc * 32], 64, lane, 32),
                         load_B(&vLds[kc * 32][0], 16, lane), c);
        }
        _Float16* op = outbuf + ((size_t)b * DIM + head * HD + n) * NPIX;
#pragma unroll
        for (int j0 = 0; j0 < 8; ++j0) {
            int m  = j0 + (half ? 8 : 0);
            int qq = mt * 16 + m;
            op[(size_t)(y0 + (qq >> 3)) * 256 + x0 + (qq & 7)] = (_Float16)c[j0];
        }
    }
}

// ---------------------------------------------------------------------------
// Kernel 4: output projection GEMM. (524288 x 64) @ (64 x 64) -> f32 d_out.
// grid = (4096, 4), block = 256 (8 waves).
// ---------------------------------------------------------------------------
__global__ void __launch_bounds__(256)
proj_gemm_kernel(const _Float16* __restrict__ inb,
                 const float* __restrict__ w,
                 const float* __restrict__ bias,
                 float* __restrict__ out) {
    __shared__ _Float16 Bsh[64][16];
    __shared__ _Float16 Ash[8][16][64];
    int lane = threadIdx.x & 31;
    int wave = threadIdx.x >> 5;
    int o0   = blockIdx.y * 16;

    for (int idx = threadIdx.x; idx < 1024; idx += 256) {
        int k = idx >> 4, nn = idx & 15;
        Bsh[k][nn] = (_Float16)w[(o0 + nn) * DIM + k];
    }

    long pix0 = (long)blockIdx.x * 128 + wave * 16;
    int  b    = (int)(pix0 >> 16);
    int  p0   = (int)(pix0 & (NPIX - 1));
    const _Float16* ib = inb + (size_t)b * DIM * NPIX;
    __builtin_prefetch(ib + p0 + 16, 0, 0);
    for (int idx = lane; idx < 1024; idx += 32) {
        int c = idx >> 4, m = idx & 15;
        Ash[wave][m][c] = ib[(size_t)c * NPIX + p0 + m];
    }
    __syncthreads();

    v8f acc = {};
    acc = wmma_f16(load_A(&Ash[wave][0][0],  64, lane, 32),
                   load_B(&Bsh[0][0],  16, lane), acc);
    acc = wmma_f16(load_A(&Ash[wave][0][32], 64, lane, 32),
                   load_B(&Bsh[32][0], 16, lane), acc);

    int   n  = lane & 15, half = lane >> 4;
    float bn = bias[o0 + n];
    float* outp = out + ((size_t)b * DIM + o0 + n) * NPIX;
#pragma unroll
    for (int j = 0; j < 8; ++j) {
        int m = j + (half ? 8 : 0);
        outp[p0 + m] = acc[j] + bn;
    }
}

// ---------------------------------------------------------------------------
// Launch
// ---------------------------------------------------------------------------
extern "C" void kernel_launch(void* const* d_in, const int* in_sizes, int n_in,
                              void* d_out, int out_size, void* d_ws, size_t ws_size,
                              hipStream_t stream) {
    const float* x      = (const float*)d_in[0];
    const float* qkv_w  = (const float*)d_in[1];
    const float* qkv_b  = (const float*)d_in[2];
    const float* proj_w = (const float*)d_in[3];
    const float* proj_b = (const float*)d_in[4];
    const float* off_w  = (const float*)d_in[5];
    const float* off_b  = (const float*)d_in[6];
    const float* sc_w   = (const float*)d_in[7];
    const float* sc_b   = (const float*)d_in[8];
    const float* relb   = (const float*)d_in[9];
    float* out = (float*)d_out;

    // workspace partition (bytes):
    //   qkv f16 planes : 8*192*65536 * 2 = 201326592
    //   attn out f16   : 8* 64*65536 * 2 =  67108864
    //   off / sc f32   : 2 * 65536 * 4   =    524288
    char* ws = (char*)d_ws;
    _Float16* qkvbuf = (_Float16*)ws;
    _Float16* outbuf = qkvbuf + (size_t)NB * 192 * NPIX;
    float*    offbuf = (float*)(outbuf + (size_t)NB * DIM * NPIX);
    float*    scbuf  = offbuf + (size_t)NB * HEADS * 2 * 1024;

    // 1) pool + offset/scale heads
    pool_off_kernel<<<dim3(NB * WN * WN), 64, 0, stream>>>(
        x, off_w, off_b, sc_w, sc_b, offbuf, scbuf);

    // 2) qkv projection (M=524288, N=192, K=64)
    qkv_gemm_kernel<<<dim3(4096, 12), 256, 0, stream>>>(x, qkv_w, qkv_b, qkvbuf);

    // 3) deformable window attention, one wave per (window, head)
    attn_kernel<<<dim3(NB * HEADS * WN * WN), 32, 0, stream>>>(
        qkvbuf, offbuf, scbuf, relb, outbuf);

    // 4) output projection (M=524288, N=64, K=64) -> f32
    proj_gemm_kernel<<<dim3(4096, 4), 256, 0, stream>>>(outbuf, proj_w, proj_b, out);
}
